// DecoderLSTM_36129264894188
// MI455X (gfx1250) — matrix operand order repaired and checked by reading, not compiled, not run on hardware
//
#include <hip/hip_runtime.h>

typedef __attribute__((ext_vector_type(16))) _Float16 v16h;
typedef __attribute__((ext_vector_type(8)))  _Float16 v8h;
typedef __attribute__((ext_vector_type(8)))  float    v8f;

#define B_    128
#define P_    196
#define ENC_  512
#define DEC_  512
#define ATT_  256
#define EMB_  256
#define V_    5000
#define VPAD_ 5120   /* padded vocab, multiple of 64 (4 N-tiles per wave) */
#define S_    256
#define S1_   257
#define GK_   1280   /* EMB + ENC + DEC : stacked [x|h] */
#define G4_   2048   /* 4*DEC */

// ---------------------------------------------------------------------------
// Pack a row-major f32 weight [Ktot x Nreal] (optionally stacked from two
// sources at row K1) into f16 WMMA B-fragment order:
//   dst[(((nTile*kChunks + kc)*32 + lane)*16 + j]
//     = B[kc*32 + (lane>>4)*16 + j][nTile*16 + (lane&15)]
// so each lane's 16 values per (kc, nTile) block are contiguous (2x b128).
// ---------------------------------------------------------------------------
__global__ void pack_wmma_b(const float* __restrict__ s1, int K1,
                            const float* __restrict__ s2,
                            int Ktot, int Nreal, int Npad,
                            _Float16* __restrict__ dst)
{
    int i = blockIdx.x * blockDim.x + threadIdx.x;
    if (i >= Ktot * Npad) return;
    const int kChunks = Ktot >> 5;
    const int j    = i & 15;
    const int lane = (i >> 4) & 31;
    const int rest = i >> 9;
    const int kc    = rest % kChunks;
    const int nTile = rest / kChunks;
    const int k = (kc << 5) + ((lane >> 4) << 4) + j;
    const int n = (nTile << 4) + (lane & 15);
    float v = 0.0f;
    if (n < Nreal)
        v = (k < K1) ? s1[(size_t)k * Nreal + n] : s2[(size_t)(k - K1) * Nreal + n];
    dst[i] = (_Float16)v;
}

__device__ __forceinline__ v16h loadB16(const _Float16* p)
{
    const v8h lo = *(const v8h*)p;
    const v8h hi = *(const v8h*)(p + 8);
    v16h r;
    #pragma unroll
    for (int j = 0; j < 8; ++j) { r[j] = lo[j]; r[8 + j] = hi[j]; }
    return r;
}

// ---------------------------------------------------------------------------
// WMMA GEMM:  C[M,Npad] = A(f32)[M,K] x Bpacked(f16)[K,Npad] + bias
// Block = 4 waves sharing one 16-row M-tile; each wave owns a 16x64 N-strip.
// The A tile is staged ONCE per block into LDS, converted to f16 and
// pre-swizzled into WMMA A-fragment order, so the inner loop is pure
// ds_load_b128 (A) + global_load_b128 (B) + 4x v_wmma per K-chunk.
// 16-bit A 16x32 / B 32x16 both in contiguous k order (k = 16*(lane/16) + s); f32 C/D per CDNA5 ISA 7.12.2.
// ---------------------------------------------------------------------------
__global__ void gemm_f32xf16(const float* __restrict__ A, int lda,
                             const _Float16* __restrict__ Bp,
                             const float* __restrict__ bias,
                             float* __restrict__ C, unsigned long long ldc,
                             int M, int Npad, int K, int Nreal)
{
    extern __shared__ _Float16 Alds[];            // [K/32][32 lanes][16] = 32*K bytes
    const int tid  = threadIdx.x;
    const int lane = tid & 31;
    const int w    = tid >> 5;                    // wave in block: 0..3
    const int grpPerRow = Npad >> 6;              // N-groups per M-tile row (multiple of 1)
    const int wm = blockIdx.x / grpPerRow;        // shared M-tile for whole block
    const int wn = (blockIdx.x - wm * grpPerRow) * 4 + 0;  // base handled per wave below
    const int kChunks = K >> 5;

    // ---- cooperative A-tile stage: global f32 -> LDS f16 in fragment order ----
    const float* Atile = A + (size_t)(wm << 4) * (size_t)lda;
    for (int m = 0; m < 16; ++m) {
        for (int k = tid; k < K; k += 128) {      // coalesced along k
            const float v = Atile[(size_t)m * lda + k];
            const int kc   = k >> 5;
            const int r    = k & 31;
            const int half = (r >> 4) & 1;        // K-run pattern of A fragment
            const int t    = r - (half << 4);
            const int j    = t;
            Alds[(kc << 9) + (((half << 4) + m) << 4) + j] = (_Float16)v;
        }
    }
    __syncthreads();

    // this wave's packed-B base (one of the 4 N-strips of the group), lane-offset
    const int myN = wn;                            // group base
    const _Float16* Bl = Bp + (((size_t)(myN + w) * 4 * 0 +
                                (size_t)((blockIdx.x % grpPerRow) * 16 + w * 4)) * kChunks << 9)
                            + ((size_t)lane << 4);

    v8f acc[4] = {};
    for (int kc = 0; kc < kChunks; ++kc) {
        const v16h a = *(const v16h*)(Alds + (kc << 9) + (lane << 4));  // 2x ds_load_b128
        const v16h b0 = loadB16(Bl + ((size_t)(0 * kChunks + kc) << 9));
        const v16h b1 = loadB16(Bl + ((size_t)(1 * kChunks + kc) << 9));
        const v16h b2 = loadB16(Bl + ((size_t)(2 * kChunks + kc) << 9));
        const v16h b3 = loadB16(Bl + ((size_t)(3 * kChunks + kc) << 9));
        acc[0] = __builtin_amdgcn_wmma_f32_16x16x32_f16(false, a, false, b0,
                                                        (short)0, acc[0], false, false);
        acc[1] = __builtin_amdgcn_wmma_f32_16x16x32_f16(false, a, false, b1,
                                                        (short)0, acc[1], false, false);
        acc[2] = __builtin_amdgcn_wmma_f32_16x16x32_f16(false, a, false, b2,
                                                        (short)0, acc[2], false, false);
        acc[3] = __builtin_amdgcn_wmma_f32_16x16x32_f16(false, a, false, b3,
                                                        (short)0, acc[3], false, false);
        // speculative prefetch of next chunk's first B block (dropped if OOB)
        __builtin_prefetch((const void*)(Bl + ((size_t)(kc + 1) << 9)), 0, 3);
    }

    const int tileN0 = (blockIdx.x % grpPerRow) * 16 + w * 4;   // in 16-col tiles
    #pragma unroll
    for (int tn = 0; tn < 4; ++tn) {
        const int ncol = ((tileN0 + tn) << 4) + (lane & 15);
        if (ncol < Nreal) {                        // guard keeps vocab padding out of d_out
            const float bv = bias ? bias[ncol] : 0.0f;
            #pragma unroll
            for (int r = 0; r < 8; ++r) {
                const int mm = (wm << 4) + r + ((lane >> 4) << 3);
                C[(size_t)mm * ldc + ncol] = acc[tn][r] + bv;
            }
        }
    }
}

// ---------------------------------------------------------------------------
// One-time helpers
// ---------------------------------------------------------------------------
__global__ void add_vec(const float* __restrict__ a, const float* __restrict__ b,
                        float* __restrict__ d, int n)
{
    int i = blockIdx.x * blockDim.x + threadIdx.x;
    if (i < n) d[i] = a[i] + b[i];
}

__global__ void mean_features(const float* __restrict__ f, float* __restrict__ mf)
{
    int b = blockIdx.x, t = threadIdx.x;
    for (int e = t; e < ENC_; e += 256) {
        float s = 0.0f;
        const float* fb = f + (size_t)b * P_ * ENC_ + e;
        for (int p = 0; p < P_; ++p) s += fb[(size_t)p * ENC_];
        mf[b * ENC_ + e] = s * (1.0f / (float)P_);
    }
}

__global__ void copy_h_to_xh(const float* __restrict__ h, float* __restrict__ xh)
{
    int i = blockIdx.x * blockDim.x + threadIdx.x;
    if (i >= B_ * DEC_) return;
    int b = i >> 9, d = i & (DEC_ - 1);
    xh[b * GK_ + EMB_ + ENC_ + d] = h[i];
}

// ---------------------------------------------------------------------------
// Per-step kernels
// ---------------------------------------------------------------------------
// scores[b,p] = va_b + sum_a tanh(Uf[b,p,a] + hWa[b,a]) * va[a]   (one wave per row)
__global__ void attn_scores(const float* __restrict__ Uf, const float* __restrict__ hWa,
                            const float* __restrict__ va, const float* __restrict__ va_b,
                            float* __restrict__ scores)
{
    int gid  = blockIdx.x * blockDim.x + threadIdx.x;
    int wave = gid >> 5, lane = gid & 31;
    if (wave >= B_ * P_) return;
    int b = wave / P_;
    const float* u  = Uf  + (size_t)wave * ATT_;
    const float* hw = hWa + (size_t)b * ATT_;
    float acc = 0.0f;
    #pragma unroll 4
    for (int a = lane; a < ATT_; a += 32)
        acc += tanhf(u[a] + hw[a]) * va[a];
    for (int off = 16; off > 0; off >>= 1)
        acc += __shfl_down(acc, off, 32);
    if (lane == 0) scores[wave] = acc + va_b[0];
}

// softmax over P, context = alpha^T features, gather embedding -> xh[b] = [emb|ctx|(h)]
__global__ void softmax_ctx_embed(const float* __restrict__ scores,
                                  const float* __restrict__ features,
                                  const int* __restrict__ captions,
                                  const float* __restrict__ emb,
                                  float* __restrict__ xh, int t)
{
    __shared__ float sal[P_];
    __shared__ float red[256];
    int b = blockIdx.x, tid = threadIdx.x;

    float v = (tid < P_) ? scores[b * P_ + tid] : -3.402823466e38f;
    red[tid] = v;
    __syncthreads();
    for (int s = 128; s > 0; s >>= 1) {
        if (tid < s) red[tid] = fmaxf(red[tid], red[tid + s]);
        __syncthreads();
    }
    float mx = red[0];
    __syncthreads();
    float e = (tid < P_) ? __expf(v - mx) : 0.0f;
    red[tid] = e;
    __syncthreads();
    for (int s = 128; s > 0; s >>= 1) {
        if (tid < s) red[tid] += red[tid + s];
        __syncthreads();
    }
    float inv = 1.0f / red[0];
    if (tid < P_) sal[tid] = e * inv;
    __syncthreads();

    const float* fb = features + (size_t)b * P_ * ENC_;
    for (int ee = tid; ee < ENC_; ee += 256) {
        float acc = 0.0f;
        for (int p = 0; p < P_; ++p) acc += sal[p] * fb[(size_t)p * ENC_ + ee];
        xh[b * GK_ + EMB_ + ee] = acc;
    }
    int cap = captions[b * S1_ + t];
    if (tid < EMB_) xh[b * GK_ + tid] = emb[(size_t)cap * EMB_ + tid];
}

// c = sig(f)*c + sig(i)*tanh(g); h = sig(o)*tanh(c); also store h into xh slot
__global__ void lstm_update(const float* __restrict__ gates, float* __restrict__ c,
                            float* __restrict__ h, float* __restrict__ xh)
{
    int i = blockIdx.x * blockDim.x + threadIdx.x;
    if (i >= B_ * DEC_) return;
    int b = i >> 9, d = i & (DEC_ - 1);
    const float* g = gates + (size_t)b * G4_;
    float ig = g[d], fg = g[DEC_ + d], gg = g[2 * DEC_ + d], og = g[3 * DEC_ + d];
    float si = 1.0f / (1.0f + __expf(-ig));
    float sf = 1.0f / (1.0f + __expf(-fg));
    float so = 1.0f / (1.0f + __expf(-og));
    float cn = sf * c[i] + si * tanhf(gg);
    float hn = so * tanhf(cn);
    c[i] = cn;
    h[i] = hn;
    xh[b * GK_ + EMB_ + ENC_ + d] = hn;
}

// ---------------------------------------------------------------------------
extern "C" void kernel_launch(void* const* d_in, const int* in_sizes, int n_in,
                              void* d_out, int out_size, void* d_ws, size_t ws_size,
                              hipStream_t stream)
{
    const float* features = (const float*)d_in[0];
    const int*   captions = (const int*)  d_in[1];
    const float* emb      = (const float*)d_in[2];
    const float* Ua_w     = (const float*)d_in[3];
    const float* Ua_b     = (const float*)d_in[4];
    const float* Wa_w     = (const float*)d_in[5];
    const float* Wa_b     = (const float*)d_in[6];
    const float* va_w     = (const float*)d_in[7];
    const float* va_b     = (const float*)d_in[8];
    const float* ih_w     = (const float*)d_in[9];
    const float* ih_b     = (const float*)d_in[10];
    const float* ic_w     = (const float*)d_in[11];
    const float* ic_b     = (const float*)d_in[12];
    const float* W_ih     = (const float*)d_in[13];
    const float* b_ih     = (const float*)d_in[14];
    const float* W_hh     = (const float*)d_in[15];
    const float* b_hh     = (const float*)d_in[16];
    const float* fcn_w    = (const float*)d_in[17];
    const float* fcn_b    = (const float*)d_in[18];
    float* out = (float*)d_out;

    // ---- workspace carve-up (all offsets 256B aligned; ~42 MB total) ----
    char* ws = (char*)d_ws;
    size_t off = 0;
    auto take = [&](size_t bytes) -> char* {
        char* p = ws + off;
        off += (bytes + 255) & ~(size_t)255;
        return p;
    };
    _Float16* Ua16  = (_Float16*)take((size_t)ENC_ * ATT_ * 2);
    _Float16* Wa16  = (_Float16*)take((size_t)DEC_ * ATT_ * 2);
    _Float16* ih16  = (_Float16*)take((size_t)ENC_ * DEC_ * 2);
    _Float16* ic16  = (_Float16*)take((size_t)ENC_ * DEC_ * 2);
    _Float16* Wg16  = (_Float16*)take((size_t)GK_  * G4_  * 2);   // stacked [W_ih; W_hh]
    _Float16* fcn16 = (_Float16*)take((size_t)DEC_ * VPAD_ * 2);  // vocab padded to 5120
    float* bg     = (float*)take((size_t)G4_ * 4);                // b_ih + b_hh
    float* Uf     = (float*)take((size_t)B_ * P_ * ATT_ * 4);
    float* meanf  = (float*)take((size_t)B_ * ENC_ * 4);
    float* h      = (float*)take((size_t)B_ * DEC_ * 4);
    float* c      = (float*)take((size_t)B_ * DEC_ * 4);
    float* hWa    = (float*)take((size_t)B_ * ATT_ * 4);
    float* scores = (float*)take((size_t)B_ * P_ * 4);
    float* xh     = (float*)take((size_t)B_ * GK_ * 4);           // [emb | ctx | h]
    float* gates  = (float*)take((size_t)B_ * G4_ * 4);

    const int CT = 256;
    auto cdiv = [](int a, int b) { return (a + b - 1) / b; };
    auto gemm = [&](const float* A, int lda, const _Float16* Bp,
                    const float* bias, float* C, unsigned long long ldc,
                    int M, int Npad, int K, int Nreal) {
        int blocks = (M / 16) * (Npad / 256);       // block = 4 waves = 16x256 of C
        size_t ldsB = (size_t)32 * K;               // f16 A tile in fragment order
        gemm_f32xf16<<<blocks, 128, ldsB, stream>>>(A, lda, Bp, bias, C, ldc,
                                                    M, Npad, K, Nreal);
    };
    auto pack = [&](const float* s1, int K1, const float* s2, int Ktot,
                    int Nreal, int Npad, _Float16* dst) {
        pack_wmma_b<<<cdiv(Ktot * Npad, CT), CT, 0, stream>>>(s1, K1, s2, Ktot,
                                                              Nreal, Npad, dst);
    };

    // ---- one-time: weight packing, mean, h0/c0, Uf ----
    pack(Ua_w, ENC_, nullptr, ENC_, ATT_, ATT_, Ua16);
    pack(Wa_w, DEC_, nullptr, DEC_, ATT_, ATT_, Wa16);
    pack(ih_w, ENC_, nullptr, ENC_, DEC_, DEC_, ih16);
    pack(ic_w, ENC_, nullptr, ENC_, DEC_, DEC_, ic16);
    pack(W_ih, EMB_ + ENC_, W_hh, GK_, G4_, G4_, Wg16);           // stacked gate weights
    pack(fcn_w, DEC_, nullptr, DEC_, V_, VPAD_, fcn16);           // padded vocab
    add_vec<<<cdiv(G4_, CT), CT, 0, stream>>>(b_ih, b_hh, bg, G4_);

    mean_features<<<B_, 256, 0, stream>>>(features, meanf);
    gemm(meanf, ENC_, ih16, ih_b, h, DEC_, B_, DEC_, ENC_, DEC_);   // h0
    gemm(meanf, ENC_, ic16, ic_b, c, DEC_, B_, DEC_, ENC_, DEC_);   // c0
    copy_h_to_xh<<<cdiv(B_ * DEC_, CT), CT, 0, stream>>>(h, xh);
    // Uf = features @ Ua + Ua_b   ([B*P, ENC] x [ENC, ATT])
    gemm(features, ENC_, Ua16, Ua_b, Uf, ATT_, B_ * P_, ATT_, ENC_, ATT_);

    // ---- 256 sequential decode steps ----
    for (int t = 0; t < S_; ++t) {
        // hWa = h @ Wa + Wa_b
        gemm(h, DEC_, Wa16, Wa_b, hWa, ATT_, B_, ATT_, DEC_, ATT_);
        // attention scores (tanh + dot over ATT)
        attn_scores<<<cdiv(B_ * P_ * 32, CT), CT, 0, stream>>>(Uf, hWa, va_w, va_b, scores);
        // softmax over P, context, embedding gather -> xh[:, :EMB+ENC]
        softmax_ctx_embed<<<B_, 256, 0, stream>>>(scores, features, captions, emb, xh, t);
        // gates = [x|h] @ [W_ih;W_hh] + (b_ih + b_hh)
        gemm(xh, GK_, Wg16, bg, gates, G4_, B_, G4_, GK_, G4_);
        // LSTM pointwise update (also refreshes h slot inside xh)
        lstm_update<<<cdiv(B_ * DEC_, CT), CT, 0, stream>>>(gates, c, h, xh);
        // out[:, t, :] = h @ fcn + fcn_b   (Nreal guard keeps pad out of d_out)
        gemm(h, DEC_, fcn16, fcn_b, out + (size_t)t * V_,
             (unsigned long long)S_ * V_, B_, VPAD_, DEC_, V_);
    }
}
